// T5Attention_35777077576109
// MI455X (gfx1250) — compile-verified, matrix operand order checked by reading, not run on hardware
//
#include <hip/hip_runtime.h>

typedef _Float16 v8h  __attribute__((ext_vector_type(8)));
typedef _Float16 v16h __attribute__((ext_vector_type(16)));
typedef float    v8f  __attribute__((ext_vector_type(8)));
typedef float    v4f  __attribute__((ext_vector_type(4)));
typedef unsigned int u32x4 __attribute__((ext_vector_type(4)));
typedef int          i32x4 __attribute__((ext_vector_type(4)));
typedef int          i32x8 __attribute__((ext_vector_type(8)));

union Frag16 { v16h v; v8h h[2]; };

#define S_LEN  2048
#define DMODEL 1024
#define NHEADS 16
#define DHEAD  64
#define GM 4096   // B * S tokens
#define GN 1024
#define GK 1024

#define WMMA_F16(a, b, c) \
  __builtin_amdgcn_wmma_f32_16x16x32_f16(false, (a), false, (b), (short)0, (c), false, false)

// ---- CDNA5 async global->LDS copy (16B per lane, tracked by ASYNCcnt) ----
__device__ __forceinline__ void async_b128(void* lds, const void* g) {
  unsigned l = (unsigned)(unsigned long long)lds;  // flat LDS aperture: [31:0] = LDS offset
  unsigned long long ga = (unsigned long long)g;
  asm volatile("global_load_async_to_lds_b128 %0, %1, off" ::"v"(l), "v"(ga)
               : "memory");
}
#define WAIT_ASYNC() asm volatile("s_wait_asynccnt 0" ::: "memory")

// ---- CDNA5 Tensor Data Mover: 2D tile (16-bit elems) global -> LDS --------
// tile0 halves per row, tile1 rows, stride0 = row pitch in elements.
// LDS padding: +16B after every 128B so rows land on a 72-half (144B) pitch
// (pad_interval code 4 = 32 DWORDs, pad_amount code 3 = 4 DWORDs).
__device__ __forceinline__ void tdm_load_2d(void* lds, const void* gp,
                                            unsigned tile0, unsigned tile1,
                                            unsigned stride0) {
  const unsigned long long ga = (unsigned long long)gp;
  u32x4 g0;
  g0[0] = 1u;                                            // count=1 (valid user D#)
  g0[1] = (unsigned)(unsigned long long)lds;             // lds_addr (bytes)
  g0[2] = (unsigned)ga;                                  // global_addr[31:0]
  g0[3] = ((unsigned)(ga >> 32) & 0x1FFFFFFu) | (2u << 30);  // addr[56:32] | type=2
  i32x8 g1;
  g1[0] = (int)((1u << 16) | (1u << 20) | (4u << 22) | (3u << 25));  // 2B, pad on
  g1[1] = (int)((tile0 & 0xFFFFu) << 16);                // tensor_dim0 = tile0 (lo)
  g1[2] = (int)(((tile0 >> 16) & 0xFFFFu) | ((tile1 & 0xFFFFu) << 16));  // d0 hi|d1 lo
  g1[3] = (int)(((tile1 >> 16) & 0xFFFFu) | ((tile0 & 0xFFFFu) << 16));  // d1 hi|tile_dim0
  g1[4] = (int)(tile1 & 0xFFFFu);                        // tile_dim1 (tile_dim2=0)
  g1[5] = (int)stride0;                                  // tensor_dim0_stride[31:0]
  g1[6] = 0;                                             // stride0 hi | stride1 lo
  g1[7] = 0;
  const i32x4 z4 = {0, 0, 0, 0};
#if __clang_major__ >= 23
  const i32x8 z8 = {0, 0, 0, 0, 0, 0, 0, 0};
  __builtin_amdgcn_tensor_load_to_lds(g0, g1, z4, z4, z8, 0);
#else
  __builtin_amdgcn_tensor_load_to_lds(g0, g1, z4, z4, 0);
#endif
}

// lane-pair exchange (lanes L <-> L^16): DS_SWIZZLE group-of-32 SWAPX16
__device__ __forceinline__ float swap16(float x) {
  return __int_as_float(__builtin_amdgcn_ds_swizzle(__float_as_int(x), 0x401f));
}

__device__ __forceinline__ v8f vzero8() {
  v8f z;
  for (int e = 0; e < 8; ++e) z[e] = 0.0f;
  return z;
}

// 16-bit A/B fragment: elems 0-7 = row[kb..kb+7], elems 8-15 = row[kb+16..kb+23]
// (kb already includes the 8*(lane>=16) group offset)
__device__ __forceinline__ v16h ld_frag(const _Float16* row, int kb) {
  Frag16 f;
  f.h[0] = *(const v8h*)(row + kb);
  f.h[1] = *(const v8h*)(row + kb + 16);
  return f.v;
}

// ---------------------------------------------------------------------------
// Prep: f32 -> f16 elementwise (hidden states)
// ---------------------------------------------------------------------------
__global__ __launch_bounds__(256) void cvt_f32_f16(const float* __restrict__ x,
                                                   _Float16* __restrict__ y,
                                                   int n) {
  const int i = (blockIdx.x * 256 + threadIdx.x) * 8;
  if (i + 8 <= n) {
    v4f a = *(const v4f*)(x + i);
    v4f b = *(const v4f*)(x + i + 4);
    v8h d;
    for (int e = 0; e < 4; ++e) {
      d[e]     = (_Float16)a[e];
      d[4 + e] = (_Float16)b[e];
    }
    *(v8h*)(y + i) = d;
  }
}

// ---------------------------------------------------------------------------
// Prep: weight [K][N] f32 -> [N][K] f16 (LDS-tiled 64x64 transpose)
// ---------------------------------------------------------------------------
__global__ __launch_bounds__(256) void transpose_w(const float* __restrict__ W,
                                                   _Float16* __restrict__ Bt) {
  __shared__ __align__(16) _Float16 sT[64 * 72];
  const int t  = threadIdx.x;
  const int kt = blockIdx.y * 64, nt = blockIdx.x * 64;
  for (int c = t; c < 1024; c += 256) {  // 64x64 elems as v4f chunks
    const int k = c >> 4, nc = (c & 15) * 4;
    v4f x = *(const v4f*)(W + (size_t)(kt + k) * GN + nt + nc);
    for (int e = 0; e < 4; ++e) sT[(nc + e) * 72 + k] = (_Float16)x[e];
  }
  __syncthreads();
  for (int c = t; c < 512; c += 256) {  // write transposed rows as b128
    const int n = c >> 3, cc = (c & 7) * 8;
    *(v8h*)(Bt + (size_t)(nt + n) * GK + kt + cc) = *(const v8h*)(sT + n * 72 + cc);
  }
}

// ---------------------------------------------------------------------------
// GEMM: C[GM,GN] = A[GM,GK] * Bt[GN,GK]^T, all-f16 inputs, f32 accum.
// Workgroup 256 threads (8 waves), tile 64(M) x 256(N), BK=32, double-buffered
// LDS fed entirely by async-to-LDS b128 copies. Wave tile 32x64 = 8 WMMA/chunk.
// CMODE: 0 = f16 row-major, 1 = f32 row-major, 2 = f16 scattered as V^T[b][h][d][s]
// ---------------------------------------------------------------------------
#define BM 64
#define BN 256
#define BK 32
#define PA 40  // LDS pitch (halves); 80B, 16B-aligned
#define PB 40

template <int CMODE>
__global__ __launch_bounds__(256) void gemm_kernel(
    const _Float16* __restrict__ A, const _Float16* __restrict__ Bt,
    void* __restrict__ Cp) {
  __shared__ __align__(16) _Float16 sA[2][BM * PA];
  __shared__ __align__(16) _Float16 sB[2][BN * PB];

  const int t    = threadIdx.x;
  const int lane = t & 31, w = t >> 5;
  const int g    = lane >> 4, ln = lane & 15;
  const int m0   = blockIdx.y * BM;
  const int n0   = blockIdx.x * BN;
  const int mw   = (w >> 2) * 32;  // 2 waves in M
  const int nw   = (w & 3) * 64;   // 4 waves in N
  const int kb   = 8 * g;
  const int ar   = t >> 2, ac = (t & 3) * 8;  // A staging chunk (1/thread)

  v8f acc[2][4];
  for (int i = 0; i < 2; ++i)
    for (int j = 0; j < 4; ++j) acc[i][j] = vzero8();

  auto stage = [&](int buf, int k0) {
    async_b128(&sA[buf][ar * PA + ac], A + (size_t)(m0 + ar) * GK + k0 + ac);
    for (int i = 0; i < 4; ++i) {  // B: 256 rows x 4 chunks = 1024
      const int c = t + i * 256;
      const int n = c >> 2, cc = (c & 3) * 8;
      async_b128(&sB[buf][n * PB + cc], Bt + (size_t)(n0 + n) * GK + k0 + cc);
    }
  };

  stage(0, 0);
  WAIT_ASYNC();
  __syncthreads();

  for (int k0 = 0; k0 < GK; k0 += BK) {
    const int cur = (k0 >> 5) & 1;
    if (k0 + BK < GK) stage(cur ^ 1, k0 + BK);  // overlap copy with WMMA

    v16h a0 = ld_frag(&sA[cur][(mw + ln) * PA], kb);
    v16h a1 = ld_frag(&sA[cur][(mw + 16 + ln) * PA], kb);
    for (int j = 0; j < 4; ++j) {
      v16h b = ld_frag(&sB[cur][(nw + 16 * j + ln) * PB], kb);
      acc[0][j] = WMMA_F16(a0, b, acc[0][j]);
      acc[1][j] = WMMA_F16(a1, b, acc[1][j]);
    }
    WAIT_ASYNC();
    __syncthreads();
  }

  // C frag: row m = r + 8*(lane>=16), col n = lane&15
  for (int i = 0; i < 2; ++i)
    for (int j = 0; j < 4; ++j)
      for (int r = 0; r < 8; ++r) {
        const int row = m0 + mw + 16 * i + 8 * g + r;
        const int col = n0 + nw + 16 * j + ln;
        const float val = acc[i][j][r];
        if (CMODE == 1) {
          ((float*)Cp)[(size_t)row * GN + col] = val;
        } else if (CMODE == 0) {
          ((_Float16*)Cp)[(size_t)row * GN + col] = (_Float16)val;
        } else {  // V^T: [b][h][d][s]
          const int bb = row >> 11, s = row & (S_LEN - 1);
          const int hh = col >> 6, d = col & (DHEAD - 1);
          ((_Float16*)Cp)[(((size_t)bb * NHEADS + hh) * DHEAD + d) * S_LEN + s] =
              (_Float16)val;
        }
      }
}

// ---------------------------------------------------------------------------
// T5 relative-position bucket (bidirectional, 32 buckets, max_distance 128)
// ---------------------------------------------------------------------------
__device__ __forceinline__ int rel_bucket(int rel) {
  int b  = (rel > 0) ? 16 : 0;
  int rp = rel < 0 ? -rel : rel;
  if (rp < 8) return b + rp;
  // 8 + log(rp/8)/log(16) * 8 ;  8/log(16) = 2.88539008...
  int large = 8 + (int)(__logf((float)rp * 0.125f) * 2.8853900817779268f);
  if (large > 15) large = 15;
  return b + large;
}

// ---------------------------------------------------------------------------
// Flash attention. One (batch, head, 128-query block) per WG; 8 waves x 16 q.
// Scores computed TRANSPOSED (S^T = K_blk @ Q^T): query == lane column, so
// softmax stats need one ds_swizzle(SWAPX16) and exp'd P^T C-frags ARE the
// B-frags for O^T = V^T @ P^T. Key block = 64 -> 16 WMMA per wave per iter.
// K and V^T tiles are bulk-DMA'd by the Tensor Data Mover (wave 0 / wave 1
// issue one tensor_load_to_lds each, double-buffered, TENSORcnt-tracked;
// TDM hardware padding matches the 72-half LDS pitch). Q staged once via
// async-to-LDS b128. Rel-position bias pre-expanded into an LDS table
// indexed by rel (one ds_load_b32 per score element, zero bucket math).
// ---------------------------------------------------------------------------
#define QB   128
#define KBLK 64
#define PQ   72  // 144B pitch
#define PK   72
#define PV   72
#define TABN (S_LEN + QB)  // 2176 rel values per WG

__global__ __launch_bounds__(256) void attn_kernel(
    const _Float16* __restrict__ Q, const _Float16* __restrict__ Kg,
    const _Float16* __restrict__ Vt, const float* __restrict__ relb,
    _Float16* __restrict__ ctx) {
  __shared__ __align__(16) _Float16 sQ[QB * PQ];
  __shared__ __align__(16) _Float16 sK[2][KBLK * PK];
  __shared__ __align__(16) _Float16 sV[2][DHEAD * PV];  // [d][key]
  __shared__ float sBias[32];
  __shared__ float sTab[TABN];  // bias value for rel = idx - qb - (QB-1)

  const int t    = threadIdx.x;
  const int lane = t & 31, w = t >> 5;
  const int g    = lane >> 4, ln = lane & 15;
  const int qb   = blockIdx.x * QB;
  const int h    = blockIdx.y;
  const int b    = blockIdx.z;
  const size_t base  = ((size_t)b * S_LEN) * DMODEL + (size_t)h * DHEAD;
  const size_t vbase = (((size_t)b * NHEADS + h) * DHEAD) * S_LEN;

  if (t < 32) sBias[t] = relb[t * NHEADS + h];
  __syncthreads();
  for (int i = t; i < TABN; i += 256)  // one-time bucket expansion
    sTab[i] = sBias[rel_bucket(i - qb - (QB - 1))];

  for (int c = t; c < 1024; c += 256) {  // Q tile 128x64, staged once (async)
    const int r = c >> 3, cc = (c & 7) * 8;
    async_b128(&sQ[r * PQ + cc], Q + base + (size_t)(qb + r) * DMODEL + cc);
  }
  // TDM staging: wave 0 -> K tile (64 keys x 64 d, row pitch DMODEL),
  //              wave 1 -> V^T tile (64 d x 64 keys, row pitch S_LEN).
  auto stageKV = [&](int buf, int kblk) {
    if (w == 0)
      tdm_load_2d(&sK[buf][0], Kg + base + (size_t)kblk * DMODEL, 64, 64, DMODEL);
    else if (w == 1)
      tdm_load_2d(&sV[buf][0], Vt + vbase + kblk, 64, 64, S_LEN);
  };
  stageKV(0, 0);
  WAIT_ASYNC();
  if (w < 2) __builtin_amdgcn_s_wait_tensorcnt(0);
  __syncthreads();  // staged data + sTab visible to all waves

  const int  q_local = w * 16 + ln;
  const v16h bq0 = ld_frag(&sQ[q_local * PQ], 8 * g);
  const v16h bq1 = ld_frag(&sQ[q_local * PQ], 32 + 8 * g);
  const int  q_g   = qb + q_local;
  const int  ibase = (QB - 1) + 8 * g - q_local;  // sTab idx = kblk+16s+r+ibase

  float m_run = -1e30f, l_run = 0.0f;
  v8f acc[4];
  for (int tt = 0; tt < 4; ++tt) acc[tt] = vzero8();

  for (int kblk = 0; kblk < S_LEN; kblk += KBLK) {
    const int cur = (kblk >> 6) & 1;
    if (kblk + KBLK < S_LEN) stageKV(cur ^ 1, kblk + KBLK);  // DMA next block

    // S^T: 4 key sub-tiles x (dk=64 -> 2 chunks) = 8 WMMA
    v8f st[4];
    for (int s = 0; s < 4; ++s) {
      st[s] = vzero8();
      v16h ak = ld_frag(&sK[cur][(16 * s + ln) * PK], 8 * g);
      st[s] = WMMA_F16(ak, bq0, st[s]);
      ak = ld_frag(&sK[cur][(16 * s + ln) * PK], 32 + 8 * g);
      st[s] = WMMA_F16(ak, bq1, st[s]);
    }

    // bias (table lookup) + running max
    float lm = -1e30f;
    for (int s = 0; s < 4; ++s) {
      const int ib = kblk + 16 * s + ibase;
      for (int r = 0; r < 8; ++r) {
        st[s][r] += sTab[ib + r];
        lm = fmaxf(lm, st[s][r]);
      }
    }
    lm = fmaxf(lm, swap16(lm));
    const float new_m = fmaxf(m_run, lm);
    const float corr  = __expf(m_run - new_m);
    m_run = new_m;

    // exp + pack P^T directly into B-fragment layout (keys 0-31, 32-63)
    Frag16 pb01, pb23;
    float  ls = 0.0f;
    for (int r = 0; r < 8; ++r) {
      const float p0 = __expf(st[0][r] - new_m);
      const float p1 = __expf(st[1][r] - new_m);
      const float p2 = __expf(st[2][r] - new_m);
      const float p3 = __expf(st[3][r] - new_m);
      ls += (p0 + p1) + (p2 + p3);
      pb01.h[0][r] = (_Float16)p0;
      pb01.h[1][r] = (_Float16)p1;
      pb23.h[0][r] = (_Float16)p2;
      pb23.h[1][r] = (_Float16)p3;
    }
    ls += swap16(ls);
    l_run = l_run * corr + ls;
    for (int tt = 0; tt < 4; ++tt)
      for (int e = 0; e < 8; ++e) acc[tt][e] *= corr;

    // O^T += V^T @ P^T : 4 d-subtiles x 2 key chunks = 8 WMMA
    for (int tt = 0; tt < 4; ++tt) {
      v16h av = ld_frag(&sV[cur][(16 * tt + ln) * PV], 8 * g);
      acc[tt] = WMMA_F16(av, pb01.v, acc[tt]);
      av = ld_frag(&sV[cur][(16 * tt + ln) * PV], 32 + 8 * g);
      acc[tt] = WMMA_F16(av, pb23.v, acc[tt]);
    }
    if (w < 2) __builtin_amdgcn_s_wait_tensorcnt(0);  // next block DMA done
    __syncthreads();
  }

  const float inv_l = 1.0f / l_run;
  for (int tt = 0; tt < 4; ++tt)
    for (int r = 0; r < 8; ++r) {
      const int d = 16 * tt + r + 8 * g;
      ctx[base + (size_t)q_g * DMODEL + d] = (_Float16)(acc[tt][r] * inv_l);
    }
}

// ---------------------------------------------------------------------------
extern "C" void kernel_launch(void* const* d_in, const int* in_sizes, int n_in,
                              void* d_out, int out_size, void* d_ws,
                              size_t ws_size, hipStream_t stream) {
  const float* X  = (const float*)d_in[0];  // [2,2048,1024]
  const float* Wq = (const float*)d_in[1];  // [1024,1024]
  const float* Wk = (const float*)d_in[2];
  const float* Wv = (const float*)d_in[3];
  const float* Wo = (const float*)d_in[4];
  const float* rb = (const float*)d_in[5];  // [32,16]
  float* out = (float*)d_out;

  const size_t MN = (size_t)GM * GN;  // 4M
  const size_t WN = (size_t)GK * GN;  // 1M
  _Float16* Xh  = (_Float16*)d_ws;    // 8 MB
  _Float16* Wqt = Xh + MN;            // 2 MB each (f16, transposed [N][K])
  _Float16* Wkt = Wqt + WN;
  _Float16* Wvt = Wkt + WN;
  _Float16* Wot = Wvt + WN;
  _Float16* Qh  = Wot + WN;           // 8 MB each
  _Float16* Kh  = Qh + MN;
  _Float16* Vt  = Kh + MN;            // V^T [b][h][d][s]
  _Float16* Ch  = Vt + MN;            // ctx f16  (total 48 MB)

  // prep: one-time conversions (all later staging is async b128 or TDM copies)
  cvt_f32_f16<<<(int)(MN / 2048), 256, 0, stream>>>(X, Xh, (int)MN);
  dim3 gt(16, 16);
  transpose_w<<<gt, 256, 0, stream>>>(Wq, Wqt);
  transpose_w<<<gt, 256, 0, stream>>>(Wk, Wkt);
  transpose_w<<<gt, 256, 0, stream>>>(Wv, Wvt);
  transpose_w<<<gt, 256, 0, stream>>>(Wo, Wot);

  dim3 gg(GN / BN, GM / BM);  // (4, 64)
  gemm_kernel<0><<<gg, 256, 0, stream>>>(Xh, Wqt, Qh);
  gemm_kernel<0><<<gg, 256, 0, stream>>>(Xh, Wkt, Kh);
  gemm_kernel<2><<<gg, 256, 0, stream>>>(Xh, Wvt, Vt);  // stores V^T per head

  dim3 ga(S_LEN / QB, NHEADS, 2);  // (16, 16, 2)
  attn_kernel<<<ga, 256, 0, stream>>>(Qh, Kh, Vt, rb, Ch);

  gemm_kernel<1><<<gg, 256, 0, stream>>>(Ch, Wot, out);
}
